// Mock158BitLinear_48421461295490
// MI455X (gfx1250) — matrix operand order repaired and checked by reading, not compiled, not run on hardware
//
#include <hip/hip_runtime.h>

// BitLinear GEMM for MI455X (gfx1250, wave32):
//   C[M,N] = x[M,K] * W[N,K]^T * scale[N] + bias[N],  M=8192, N=4096, K=4096
//
// Fast path (needs ~101 MB workspace):
//   1) one-shot convert x->bf16, W(ternary int32)->bf16 into d_ws
//   2) WMMA GEMM: global->LDS via async-to-LDS (ASYNCcnt), double-buffered,
//      v_wmma_f32_16x16x32_bf16 with fp32 accumulation.
// Fallback: single-kernel version converting inline (round-1 kernel).

typedef __attribute__((ext_vector_type(16))) __bf16 v16bf;
typedef __attribute__((ext_vector_type(8)))  __bf16 v8bf;
typedef __attribute__((ext_vector_type(4)))  __bf16 v4bf;
typedef __attribute__((ext_vector_type(8)))  float  v8f;

#define M_TOTAL 8192
#define N_TOTAL 4096
#define K_TOTAL 4096
#define BM 128
#define BN 128

// ---------------------------------------------------------------------------
// CDNA5 async-to-LDS helpers (inline asm: portable across both toolchains).
// GVS mode: lds-offset VGPR, 32-bit global byte-offset VGPR, 64-bit SGPR base.
// ---------------------------------------------------------------------------
__device__ __forceinline__ void async_copy_b128(unsigned lds_off, unsigned g_off,
                                                unsigned long long sbase) {
    asm volatile("global_load_async_to_lds_b128 %0, %1, %2"
                 :: "v"(lds_off), "v"(g_off), "s"(sbase) : "memory");
}
__device__ __forceinline__ void wait_asynccnt_le8() {
    asm volatile("s_wait_asynccnt 0x8" ::: "memory");
}
__device__ __forceinline__ void wait_asynccnt_0() {
    asm volatile("s_wait_asynccnt 0x0" ::: "memory");
}

// ---------------------------------------------------------------------------
// One-shot conversion kernels (memory-bound, run once per launch)
// ---------------------------------------------------------------------------
__global__ __launch_bounds__(256)
void convert_x_bf16(const float* __restrict__ x, unsigned short* __restrict__ o) {
    const size_t i = ((size_t)blockIdx.x * 256 + threadIdx.x) * 8;
    float4 a = *(const float4*)(x + i);
    float4 b = *(const float4*)(x + i + 4);
    v8bf v = { (__bf16)a.x, (__bf16)a.y, (__bf16)a.z, (__bf16)a.w,
               (__bf16)b.x, (__bf16)b.y, (__bf16)b.z, (__bf16)b.w };
    *(v8bf*)(o + i) = v;
}

__global__ __launch_bounds__(256)
void convert_w_bf16(const int* __restrict__ w, unsigned short* __restrict__ o) {
    const size_t i = ((size_t)blockIdx.x * 256 + threadIdx.x) * 8;
    int4 a = *(const int4*)(w + i);
    int4 b = *(const int4*)(w + i + 4);
    v8bf v = { (__bf16)(float)a.x, (__bf16)(float)a.y,
               (__bf16)(float)a.z, (__bf16)(float)a.w,
               (__bf16)(float)b.x, (__bf16)(float)b.y,
               (__bf16)(float)b.z, (__bf16)(float)b.w };
    *(v8bf*)(o + i) = v;
}

// ---------------------------------------------------------------------------
// Fast GEMM: bf16 inputs, async global->LDS, double-buffered, KT=64
// ---------------------------------------------------------------------------
#define KT2 64
#define LDT 72   // padded row stride (elements): 144 B = 9*16 B, keeps b128 aligned

__global__ __launch_bounds__(256, 2)
void bitlinear_wmma_bf16_async(const unsigned short* __restrict__ Ab16,
                               const unsigned short* __restrict__ Wb16,
                               const float* __restrict__ wscale,
                               const float* __restrict__ bias,
                               float*       __restrict__ out)
{
    __shared__ __bf16 ldsA[2][BM * LDT];
    __shared__ __bf16 ldsB[2][BN * LDT];

    const int tid  = threadIdx.x;
    const int lane = tid & 31;
    const int wave = tid >> 5;          // 0..7
    const int wm   = wave >> 2;         // 0..1 : 64-row wave tile
    const int wn   = wave & 3;          // 0..3 : 32-col wave tile
    const int half = lane >> 4;
    const int l16  = lane & 15;

    const int bn0 = blockIdx.x * BN;
    const int bm0 = blockIdx.y * BM;

    const unsigned long long sbaseA =
        (unsigned long long)(size_t)(Ab16 + (size_t)bm0 * K_TOTAL);
    const unsigned long long sbaseW =
        (unsigned long long)(size_t)(Wb16 + (size_t)bn0 * K_TOTAL);

    // LDS byte offsets (low 32 bits of flat shared address == LDS offset)
    unsigned ldsAb[2], ldsBb[2];
    ldsAb[0] = (unsigned)(size_t)&ldsA[0][0];
    ldsAb[1] = (unsigned)(size_t)&ldsA[1][0];
    ldsBb[0] = (unsigned)(size_t)&ldsB[0][0];
    ldsBb[1] = (unsigned)(size_t)&ldsB[1][0];

    // Stage copy mapping: tile = 128 rows x 8 16B-chunks (KT2*2 = 128 B/row).
    // 256 threads x 4 iters x (1 A + 1 B) = 8 async b128 per thread per stage.
    const int row0 = tid >> 3;            // 0..31, +32 per iter
    const unsigned c16 = (unsigned)(tid & 7) * 16u;

    auto issue_stage = [&](int buf, int k0) {
        const unsigned kb = (unsigned)k0 * 2u;
#pragma unroll
        for (int i = 0; i < 4; ++i) {
            const int row = row0 + 32 * i;
            const unsigned go = (unsigned)row * (K_TOTAL * 2u) + kb + c16;
            const unsigned lo = (unsigned)row * (LDT * 2u) + c16;
            async_copy_b128(ldsAb[buf] + lo, go, sbaseA);
            async_copy_b128(ldsBb[buf] + lo, go, sbaseW);
        }
    };

    const v8f vzero = {0.f, 0.f, 0.f, 0.f, 0.f, 0.f, 0.f, 0.f};
    v8f acc[4][2];
#pragma unroll
    for (int mt = 0; mt < 4; ++mt)
#pragma unroll
        for (int nt = 0; nt < 2; ++nt)
            acc[mt][nt] = vzero;

    auto compute_stage = [&](int buf) {
        const __bf16* As = &ldsA[buf][0];
        const __bf16* Bs = &ldsB[buf][0];
#pragma unroll
        for (int ks = 0; ks < 2; ++ks) {           // two 16x16x32 k-slices
            const int k0 = ks * 32;
            // A frags: lanes 0-15 K=0..7,16..23 ; lanes 16-31 K=8..15,24..31
            v16bf af[4];
#pragma unroll
            for (int mt = 0; mt < 4; ++mt) {
                const __bf16* base = As + (wm * 64 + mt * 16 + l16) * LDT + k0;
                v8bf lo = *(const v8bf*)(base + half * 8);
                v8bf hi = *(const v8bf*)(base + 16 + half * 8);
#pragma unroll
                for (int i = 0; i < 8; ++i) { af[mt][i] = lo[i]; af[mt][i + 8] = hi[i]; }
            }
            // B frags: lane n = l16; lanes 0-15 K=0..7,16..23 ; lanes 16-31 K=8..15,24..31
            v16bf bfrg[2];
#pragma unroll
            for (int nt = 0; nt < 2; ++nt) {
                const __bf16* base = Bs + (wn * 32 + nt * 16 + l16) * LDT + k0;
                v8bf lo = *(const v8bf*)(base + half * 8);
                v8bf hi = *(const v8bf*)(base + 16 + half * 8);
#pragma unroll
                for (int i = 0; i < 8; ++i) { bfrg[nt][i] = lo[i]; bfrg[nt][i + 8] = hi[i]; }
            }
#pragma unroll
            for (int mt = 0; mt < 4; ++mt)
#pragma unroll
                for (int nt = 0; nt < 2; ++nt)
                    acc[mt][nt] = __builtin_amdgcn_wmma_f32_16x16x32_bf16(
                        false, af[mt], false, bfrg[nt],
                        (short)0, acc[mt][nt], false, false);
        }
    };

    issue_stage(0, 0);
    const int NK = K_TOTAL / KT2;                  // 64 stages
    for (int kt = 0; kt < NK; ++kt) {
        const int cur = kt & 1;
        if (kt + 1 < NK) {
            issue_stage(cur ^ 1, (kt + 1) * KT2);  // prefetch next buffer
            wait_asynccnt_le8();                   // wait only for current buffer
        } else {
            wait_asynccnt_0();
        }
        __syncthreads();                           // all waves' cur data in LDS
        compute_stage(cur);
        __syncthreads();                           // safe to overwrite cur next iter
    }

    // Epilogue: C layout: lane%16 = N col, VGPR i = row i + 8*half
#pragma unroll
    for (int nt = 0; nt < 2; ++nt) {
        const int n = bn0 + wn * 32 + nt * 16 + l16;
        const float s = wscale[n];
        const float b = bias[n];
#pragma unroll
        for (int mt = 0; mt < 4; ++mt) {
            const size_t mbase = (size_t)bm0 + wm * 64 + mt * 16 + half * 8;
#pragma unroll
            for (int i = 0; i < 8; ++i)
                out[(mbase + i) * (size_t)N_TOTAL + n] = acc[mt][nt][i] * s + b;
        }
    }
}

// ---------------------------------------------------------------------------
// Fallback: inline-convert version (no workspace needed)
// ---------------------------------------------------------------------------
#define KT 32
#define LDA 48
#define LDB 48

__global__ __launch_bounds__(256, 2)
void bitlinear_wmma_bf16(const float* __restrict__ x,
                         const int*   __restrict__ wq,
                         const float* __restrict__ wscale,
                         const float* __restrict__ bias,
                         float*       __restrict__ out)
{
    __shared__ __bf16 ldsA[BM * LDA];
    __shared__ __bf16 ldsB[BN * LDB];

    const int tid  = threadIdx.x;
    const int lane = tid & 31;
    const int wave = tid >> 5;
    const int wm   = wave >> 2;
    const int wn   = wave & 3;
    const int half = lane >> 4;
    const int l16  = lane & 15;

    const int bn0 = blockIdx.x * BN;
    const int bm0 = blockIdx.y * BM;

    const int srow = tid >> 3;
    const int scol = (tid & 7) * 4;

    const float* xg = x  + (size_t)bm0 * K_TOTAL;
    const int*   wg = wq + (size_t)bn0 * K_TOTAL;

    float4 xa[4];
    int4   wb[4];

    auto load_tiles = [&](int k0) {
#pragma unroll
        for (int p = 0; p < 4; ++p) {
            const int row = p * 32 + srow;
            xa[p] = *(const float4*)(xg + (size_t)row * K_TOTAL + k0 + scol);
            wb[p] = *(const int4*)  (wg + (size_t)row * K_TOTAL + k0 + scol);
        }
    };

    const v8f vzero = {0.f, 0.f, 0.f, 0.f, 0.f, 0.f, 0.f, 0.f};
    v8f acc[4][2];
#pragma unroll
    for (int mt = 0; mt < 4; ++mt)
#pragma unroll
        for (int nt = 0; nt < 2; ++nt)
            acc[mt][nt] = vzero;

    load_tiles(0);

    for (int kt = 0; kt < K_TOTAL / KT; ++kt) {
        __syncthreads();
#pragma unroll
        for (int p = 0; p < 4; ++p) {
            const int row = p * 32 + srow;
            v4bf va = { (__bf16)xa[p].x, (__bf16)xa[p].y,
                        (__bf16)xa[p].z, (__bf16)xa[p].w };
            *(v4bf*)&ldsA[row * LDA + scol] = va;
            v4bf vb = { (__bf16)(float)wb[p].x, (__bf16)(float)wb[p].y,
                        (__bf16)(float)wb[p].z, (__bf16)(float)wb[p].w };
            *(v4bf*)&ldsB[row * LDB + scol] = vb;
        }
        __syncthreads();

        if (kt + 1 < K_TOTAL / KT) load_tiles((kt + 1) * KT);

        v16bf af[4];
#pragma unroll
        for (int mt = 0; mt < 4; ++mt) {
            const __bf16* base = &ldsA[(wm * 64 + mt * 16 + l16) * LDA];
            v8bf lo = *(const v8bf*)(base + half * 8);
            v8bf hi = *(const v8bf*)(base + 16 + half * 8);
#pragma unroll
            for (int i = 0; i < 8; ++i) { af[mt][i] = lo[i]; af[mt][i + 8] = hi[i]; }
        }
        v16bf bfr[2];
#pragma unroll
        for (int nt = 0; nt < 2; ++nt) {
            const __bf16* base = &ldsB[(wn * 32 + nt * 16 + l16) * LDB];
            v8bf lo = *(const v8bf*)(base + half * 8);
            v8bf hi = *(const v8bf*)(base + 16 + half * 8);
#pragma unroll
            for (int i = 0; i < 8; ++i) { bfr[nt][i] = lo[i]; bfr[nt][i + 8] = hi[i]; }
        }
#pragma unroll
        for (int mt = 0; mt < 4; ++mt)
#pragma unroll
            for (int nt = 0; nt < 2; ++nt)
                acc[mt][nt] = __builtin_amdgcn_wmma_f32_16x16x32_bf16(
                    false, af[mt], false, bfr[nt], (short)0, acc[mt][nt], false, false);
    }

#pragma unroll
    for (int nt = 0; nt < 2; ++nt) {
        const int n = bn0 + wn * 32 + nt * 16 + l16;
        const float s = wscale[n];
        const float b = bias[n];
#pragma unroll
        for (int mt = 0; mt < 4; ++mt) {
            const size_t mbase = (size_t)bm0 + wm * 64 + mt * 16 + half * 8;
#pragma unroll
            for (int i = 0; i < 8; ++i)
                out[(mbase + i) * (size_t)N_TOTAL + n] = acc[mt][nt][i] * s + b;
        }
    }
}

// ---------------------------------------------------------------------------
extern "C" void kernel_launch(void* const* d_in, const int* in_sizes, int n_in,
                              void* d_out, int out_size, void* d_ws, size_t ws_size,
                              hipStream_t stream) {
    (void)in_sizes; (void)n_in; (void)out_size;
    const float* x      = (const float*)d_in[0];   // [4, 2048, 4096] fp32
    const int*   wq     = (const int*)  d_in[1];   // [4096, 4096] int32 {-1,0,1}
    const float* wscale = (const float*)d_in[2];   // [4096, 1] fp32
    const float* bias   = (const float*)d_in[3];   // [4096] fp32
    float*       out    = (float*)d_out;           // [4, 2048, 4096] fp32

    const size_t nA = (size_t)M_TOTAL * K_TOTAL;   // 33.6M elements
    const size_t nW = (size_t)N_TOTAL * K_TOTAL;   // 16.8M elements
    dim3 grid(N_TOTAL / BN, M_TOTAL / BM);         // 32 x 64 blocks

    if (ws_size >= (nA + nW) * sizeof(unsigned short)) {
        unsigned short* wsA = (unsigned short*)d_ws;
        unsigned short* wsW = wsA + nA;
        convert_x_bf16<<<(unsigned)(nA / 2048), 256, 0, stream>>>(x, wsA);
        convert_w_bf16<<<(unsigned)(nW / 2048), 256, 0, stream>>>(wq, wsW);
        bitlinear_wmma_bf16_async<<<grid, 256, 0, stream>>>(wsA, wsW, wscale, bias, out);
    } else {
        bitlinear_wmma_bf16<<<grid, 256, 0, stream>>>(x, wq, wscale, bias, out);
    }
}
